// NMMO3EncoderRef_84937273245751
// MI455X (gfx1250) — matrix operand order repaired and checked by reading, not compiled, not run on hardware
//
#include <hip/hip_runtime.h>
#include <hip/hip_bf16.h>

typedef __attribute__((ext_vector_type(16))) _Float16 v16h;
typedef __attribute__((ext_vector_type(8)))  float    v8f;
typedef __attribute__((ext_vector_type(4)))  int      v4i;

#define B_TOTAL 8192
#define OBS_W   1707
#define K1CH    47     // conv1 K chunks of 32 (59*25=1475 -> 1504)
#define K2CH    36     // conv2 K = 1152
#define K3CH    57     // proj  K (1817 -> 1824)
#define K3P     1824
#define NPROJ   512
#define SPB     8      // samples per k_encode workgroup

// pre-swizzled fp16 weight panels: [chunk][col][40] halves (frag = contiguous 32B)
#define PWS_B   (unsigned)(K3CH*512*40*2)   // 2,334,720
#define W1S_B   (unsigned)(K1CH*128*40*2)   //   481,280
#define W2S_B   (unsigned)(K2CH*128*40*2)   //   368,640
#define OFF_PWS 0u
#define OFF_W1S (OFF_PWS + PWS_B)
#define OFF_W2S (OFF_W1S + W1S_B)
#define OFF_EMB (OFF_W2S + W2S_B)
#define OFF_CAT (OFF_EMB + 8192u)           // obs_cat fp16 [8192][1824]

__constant__ int d_offc[10] = {0,4,8,25,30,33,38,43,48,55};

// ---------------- async global->LDS (gfx1250 ASYNCcnt path), guarded --------
#if __has_builtin(__builtin_amdgcn_global_load_async_to_lds_b128) && \
    __has_builtin(__builtin_amdgcn_s_wait_asynccnt)
#define HAVE_ASYNC 1
typedef __attribute__((address_space(1))) v4i gv4i;   // global int4
typedef __attribute__((address_space(3))) v4i lv4i;   // LDS int4
#define ASYNC_WAIT(n) __builtin_amdgcn_s_wait_asynccnt(n)
#else
#define HAVE_ASYNC 0
#define ASYNC_WAIT(n)
#endif

__device__ __forceinline__ void g2l_b128(void* lds, const void* gsrc) {
#if HAVE_ASYNC
  __builtin_amdgcn_global_load_async_to_lds_b128(
      (gv4i*)(void*)gsrc, (lv4i*)lds, 0, 0);
#else
  *(v4i*)lds = *(const v4i*)gsrc;
#endif
}

// ---------------------------------------------------------------------------
// K0: build fp16 weight panels, pre-swizzled [chunk][col][40], zero-padded.
// ---------------------------------------------------------------------------
__global__ void k_prep(const float* __restrict__ conv1_w, const float* __restrict__ conv2_w,
                       const float* __restrict__ embed_w, const float* __restrict__ proj_w,
                       _Float16* __restrict__ pws, _Float16* __restrict__ w1s,
                       _Float16* __restrict__ w2s, _Float16* __restrict__ emb)
{
  const int PW_N = K3CH*512*40, W1_N = K1CH*128*40, W2_N = K2CH*128*40, EM_N = 128*32;
  const int total = PW_N + W1_N + W2_N + EM_N;
  for (int idx = blockIdx.x*blockDim.x + threadIdx.x; idx < total;
       idx += gridDim.x*blockDim.x) {
    if (idx < PW_N) {                              // pws[q][n][kk] = proj_w[n][q*32+kk]
      int q = idx / 20480, rem = idx - q*20480, n = rem/40, kk = rem - n*40;
      int k = q*32 + kk;
      float v = (kk < 32 && k < 1817) ? proj_w[(size_t)n*1817 + k] : 0.f;
      pws[idx] = (_Float16)v;
    } else if (idx < PW_N + W1_N) {                // w1s[q][o][kk], k=(c*5+dy)*5+dx
      int i2 = idx - PW_N;
      int q = i2 / 5120, rem = i2 - q*5120, o = rem/40, kk = rem - o*40;
      int k = q*32 + kk;
      float v = 0.f;
      if (kk < 32 && k < 1475) {
        int c = k/25, r = k - c*25, dy = r/5, dx = r - dy*5;
        v = conv1_w[((o*59 + c)*5 + dy)*5 + dx];
      }
      w1s[i2] = (_Float16)v;
    } else if (idx < PW_N + W1_N + W2_N) {         // w2s[q][o][kk], k=i*9+dy*3+dx
      int i2 = idx - PW_N - W1_N;
      int q = i2 / 5120, rem = i2 - q*5120, o = rem/40, kk = rem - o*40;
      int k = q*32 + kk;
      float v = 0.f;
      if (kk < 32 && k < 1152) {
        int ci = k/9, r = k - ci*9;
        v = conv2_w[((o*128 + ci)*3 + r/3)*3 + (r - (r/3)*3)];
      }
      w2s[i2] = (_Float16)v;
    } else {
      int i2 = idx - PW_N - W1_N - W2_N;
      emb[i2] = (_Float16)embed_w[i2];
    }
  }
}

// ---------------------------------------------------------------------------
// K1: 8 samples/WG, 192 threads (6 waves).
//  A: chsel table + embed/raw cols + per-row nonzero bitmaps
//  B: conv1 WMMA, one-hot A from bitmap, double-buffered async B panels
//  C: conv2 WMMA from LDS-resident conv1 output
// ---------------------------------------------------------------------------
__global__ __launch_bounds__(192) void k_encode(
    const float* __restrict__ obs, const float* __restrict__ conv1_b,
    const float* __restrict__ conv2_b, const _Float16* __restrict__ w1s,
    const _Float16* __restrict__ w2s, const _Float16* __restrict__ emb,
    _Float16* __restrict__ obs_cat)
{
  __shared__ __align__(16) unsigned char chsel[SPB][1664];   // 13,312 B
  __shared__ __align__(16) _Float16 BsBuf[2][128*40];        // 20,480 B (dbl buf)
  __shared__ __align__(16) unsigned un[6144];                // 24,576 B, aliased:
  unsigned* bm  = un;                       // 96 rows x 48 words (conv1 bitmaps)
  _Float16* x1s = (_Float16*)un;            // then: 8 samples x 12 x 128 halves

  const int t = threadIdx.x;
  const int B0 = blockIdx.x * SPB;
  const int wv = t >> 5, lane = t & 31, r = lane & 15, hi = lane >> 4;
  const int hi8 = hi * 8;

  // issue async prefetch of conv1 B chunk 0 ASAP (overlaps phase A)
  #pragma unroll
  for (int i = 0; i < 4; ++i) {
    int idx = t + i*192; if (idx > 639) idx = 639;
    g2l_b128((char*)&BsBuf[0][0] + idx*16, (const char*)w1s + idx*16);
  }

  // ---- Phase A ------------------------------------------------------------
  for (int idx = t; idx < SPB*1664; idx += 192) {        // code->channel table
    int s = idx / 1664, j = idx - s*1664;
    unsigned char v = 0xFF;
    if (j < 1650) v = (unsigned char)((int)obs[(size_t)(B0+s)*OBS_W + j] + d_offc[j % 10]);
    chsel[s][j] = v;
  }
  for (int idx = t; idx < SPB*1568; idx += 192) {        // cols 256..1823
    int s = idx / 1568, c0 = idx - s*1568;
    size_t b = (size_t)(B0 + s);
    _Float16 hv = (_Float16)0.f;
    if (c0 < 1504) {
      int j = c0 >> 5, e = c0 & 31;
      int code = (int)obs[b*OBS_W + 1650 + j];
      hv = emb[code*32 + e];
    } else if (c0 < 1551) hv = (_Float16)obs[b*OBS_W + 1650 + (c0 - 1504)];
    else if (c0 < 1561)   hv = (_Float16)obs[b*OBS_W + 1697 + (c0 - 1551)];
    obs_cat[b*K3P + 256 + c0] = hv;
  }
  for (int idx = t; idx < 96*48; idx += 192) bm[idx] = 0u;
  __syncthreads();

  {                                                       // build row bitmaps
    int row = t >> 1, c0 = (t & 1) * 5;
    int s = row/12, p = row - s*12, y = p >> 2, x = p & 3;
    for (int c = c0; c < c0 + 5; ++c)
      #pragma unroll
      for (int dy = 0; dy < 5; ++dy)
        #pragma unroll
        for (int dx = 0; dx < 5; ++dx) {
          int ch = chsel[s][((3*y+dy)*15 + (3*x+dx))*10 + c];
          int k = ch*25 + dy*5 + dx;
          atomicOr(&bm[row*48 + (k >> 5)], 1u << (k & 31));
        }
  }
  __syncthreads();

  // ---- Phase B: conv1 GEMM, 96 rows = 6 wave tiles, K = 47 chunks ---------
  {
    v8f acc[8] = {};
    const int m = wv*16 + r;
    for (int q = 0; q < K1CH; ++q) {
      if (q + 1 < K1CH) {                                 // prefetch next chunk
        const char* src = (const char*)w1s + (size_t)(q+1)*10240;
        char* dst = (char*)&BsBuf[(q+1)&1][0];
        #pragma unroll
        for (int i = 0; i < 4; ++i) {
          int idx = t + i*192; if (idx > 639) idx = 639;
          g2l_b128(dst + idx*16, src + idx*16);
        }
        ASYNC_WAIT(4);
      } else ASYNC_WAIT(0);
      __syncthreads();
      const _Float16* Bcur = &BsBuf[q & 1][0];
      unsigned bw = bm[m*48 + q];                         // row's nonzero bits
      v16h a;
      #pragma unroll
      for (int j = 0; j < 16; ++j) {
        int kk = j + hi*16;
        a[j] = ((bw >> kk) & 1u) ? (_Float16)1.f : (_Float16)0.f;
      }
      #pragma unroll
      for (int g4 = 0; g4 < 2; ++g4) {
        v16h bf[4];
        #pragma unroll
        for (int u = 0; u < 4; ++u)
          bf[u] = *(const v16h*)&Bcur[((g4*4+u)*16 + r)*40 + hi*16];
        #pragma unroll
        for (int u = 0; u < 4; ++u)
          acc[g4*4+u] = __builtin_amdgcn_wmma_f32_16x16x32_f16(
              false, a, false, bf[u], (short)0, acc[g4*4+u], false, false);
      }
      __syncthreads();
    }
    // prefetch conv2 chunk 0 (BsBuf[0] free after final barrier above)
    #pragma unroll
    for (int i = 0; i < 4; ++i) {
      int idx = t + i*192; if (idx > 639) idx = 639;
      g2l_b128((char*)&BsBuf[0][0] + idx*16, (const char*)w2s + idx*16);
    }
    #pragma unroll
    for (int nt = 0; nt < 8; ++nt) {                      // bias+ReLU -> x1s
      int o = nt*16 + r;
      float bias = conv1_b[o];
      #pragma unroll
      for (int j = 0; j < 8; ++j) {
        int m2 = wv*16 + j + hi8;
        int s2 = m2/12, p2 = m2 - s2*12;
        float v = acc[nt][j] + bias;
        x1s[s2*1536 + p2*128 + o] = (_Float16)(v > 0.f ? v : 0.f);
      }
    }
  }
  __syncthreads();

  // ---- Phase C: conv2 GEMM, one 16-row tile (8 samples x 2 pos) -----------
  {
    v8f acc2[2] = {};
    const int sC = r >> 1, x2 = r & 1;
    for (int q = 0; q < K2CH; ++q) {
      if (q + 1 < K2CH) {
        const char* src = (const char*)w2s + (size_t)(q+1)*10240;
        char* dst = (char*)&BsBuf[(q+1)&1][0];
        #pragma unroll
        for (int i = 0; i < 4; ++i) {
          int idx = t + i*192; if (idx > 639) idx = 639;
          g2l_b128(dst + idx*16, src + idx*16);
        }
        ASYNC_WAIT(4);
      } else ASYNC_WAIT(0);
      __syncthreads();
      if (wv < 4) {
        const _Float16* Bcur = &BsBuf[q & 1][0];
        int k0 = q*32;
        v16h a;
        #pragma unroll
        for (int j = 0; j < 16; ++j) {                    // im2col gather
          int kk = j + hi*16;
          int k = k0 + kk;
          int ci = k/9, rr = k - ci*9, dy = rr/3, dx = rr - dy*3;
          a[j] = x1s[sC*1536 + (dy*4 + dx + x2)*128 + ci];
        }
        v16h bf[2];
        #pragma unroll
        for (int u = 0; u < 2; ++u)
          bf[u] = *(const v16h*)&Bcur[((wv*2+u)*16 + r)*40 + hi*16];
        #pragma unroll
        for (int u = 0; u < 2; ++u)
          acc2[u] = __builtin_amdgcn_wmma_f32_16x16x32_f16(
              false, a, false, bf[u], (short)0, acc2[u], false, false);
      }
      __syncthreads();
    }
    if (wv < 4) {
      #pragma unroll
      for (int u = 0; u < 2; ++u) {
        int o = (wv*2 + u)*16 + r;
        float bias = conv2_b[o];
        #pragma unroll
        for (int j = 0; j < 8; ++j) {
          int m2 = j + hi8;
          size_t b = (size_t)(B0 + (m2 >> 1));
          obs_cat[b*K3P + (o*2 + (m2 & 1))] = (_Float16)(acc2[u][j] + bias);
        }
      }
    }
  }
}

// ---------------------------------------------------------------------------
// K2: out = relu(obs_cat @ pwt + b). 256 threads = 8 waves (4 M x 2 N-128),
// 64 x 256 tile/WG, double-buffered async B panels, A slab frag-swizzled.
// ---------------------------------------------------------------------------
__global__ __launch_bounds__(256) void k_proj(
    const _Float16* __restrict__ obs_cat, const _Float16* __restrict__ pws,
    const float* __restrict__ proj_b, float* __restrict__ out)
{
  __shared__ __align__(16) _Float16 As[4*32*16];        //  4,096 B
  __shared__ __align__(16) _Float16 BsBuf[2][256*40];   // 40,960 B
  const int t = threadIdx.x;
  const int wv = t >> 5, lane = t & 31, r = lane & 15, hi = lane >> 4;
  const int mi = wv >> 1, ni = wv & 1;
  const int Mbase = (blockIdx.x >> 1) * 64;
  const int Nbase = (blockIdx.x & 1) * 256;

  // prefetch B chunk 0
  #pragma unroll
  for (int i = 0; i < 5; ++i) {
    int idx = t + i*256;
    g2l_b128((char*)&BsBuf[0][0] + idx*16,
             (const char*)(pws + (size_t)Nbase*40) + idx*16);
  }

  v8f acc[8] = {};
  for (int q = 0; q < K3CH; ++q) {
    if (q + 1 < K3CH) {
      const char* src = (const char*)(pws + ((size_t)(q+1)*512 + Nbase)*40);
      char* dst = (char*)&BsBuf[(q+1)&1][0];
      #pragma unroll
      for (int i = 0; i < 5; ++i) {
        int idx = t + i*256;
        g2l_b128(dst + idx*16, src + idx*16);
      }
    }
    int k0 = q*32;
    for (int idx = t; idx < 2048; idx += 256) {         // A slab -> frag order
      int r64 = idx >> 5, kk = idx & 31;
      int mt = r64 >> 4, rr = r64 & 15;
      int hiK = kk >> 4;
      int slot = kk & 15;
      As[(mt*32 + rr + hiK*16)*16 + slot] =
          obs_cat[(size_t)(Mbase + r64)*K3P + k0 + kk];
    }
    if (q + 1 < K3CH) { ASYNC_WAIT(5); } else { ASYNC_WAIT(0); }
    __syncthreads();
    const _Float16* Bcur = &BsBuf[q & 1][0];
    v16h a = *(const v16h*)&As[(mi*32 + lane)*16];
    #pragma unroll
    for (int g4 = 0; g4 < 2; ++g4) {
      v16h bf[4];
      #pragma unroll
      for (int u = 0; u < 4; ++u)
        bf[u] = *(const v16h*)&Bcur[(ni*128 + (g4*4+u)*16 + r)*40 + hi*16];
      #pragma unroll
      for (int u = 0; u < 4; ++u)
        acc[g4*4+u] = __builtin_amdgcn_wmma_f32_16x16x32_f16(
            false, a, false, bf[u], (short)0, acc[g4*4+u], false, false);
    }
    __syncthreads();
  }
  #pragma unroll
  for (int nt = 0; nt < 8; ++nt) {                      // bias + ReLU -> f32
    int n = Nbase + ni*128 + nt*16 + r;
    float bias = proj_b[n];
    #pragma unroll
    for (int j = 0; j < 8; ++j) {
      int b = Mbase + mi*16 + j + hi*8;
      float v = acc[nt][j] + bias;
      out[(size_t)b*NPROJ + n] = v > 0.f ? v : 0.f;
    }
  }
}

// ---------------------------------------------------------------------------
extern "C" void kernel_launch(void* const* d_in, const int* in_sizes, int n_in,
                              void* d_out, int out_size, void* d_ws, size_t ws_size,
                              hipStream_t stream)
{
  const float* obs     = (const float*)d_in[0];
  const float* conv1_w = (const float*)d_in[1];
  const float* conv1_b = (const float*)d_in[2];
  const float* conv2_w = (const float*)d_in[3];
  const float* conv2_b = (const float*)d_in[4];
  const float* embed_w = (const float*)d_in[5];
  const float* proj_w  = (const float*)d_in[6];
  const float* proj_b  = (const float*)d_in[7];

  char* ws = (char*)d_ws;                               // ~33 MB scratch
  _Float16* pws = (_Float16*)(ws + OFF_PWS);
  _Float16* w1s = (_Float16*)(ws + OFF_W1S);
  _Float16* w2s = (_Float16*)(ws + OFF_W2S);
  _Float16* emb = (_Float16*)(ws + OFF_EMB);
  _Float16* cat = (_Float16*)(ws + OFF_CAT);

  k_prep<<<6236, 256, 0, stream>>>(conv1_w, conv2_w, embed_w, proj_w,
                                   pws, w1s, w2s, emb);
  k_encode<<<B_TOTAL/SPB, 192, 0, stream>>>(obs, conv1_b, conv2_b,
                                            w1s, w2s, emb, cat);
  k_proj<<<(B_TOTAL/64)*2, 256, 0, stream>>>(cat, pws, proj_b, (float*)d_out);
}
